// SemlaSelfAttention_257698038062
// MI455X (gfx1250) — compile-verified, matrix operand order checked by reading, not run on hardware
//
#include <hip/hip_runtime.h>
#include <hip/hip_bf16.h>

typedef __attribute__((ext_vector_type(16))) _Float16 v16h;
typedef __attribute__((ext_vector_type(8)))  _Float16 v8h;
typedef __attribute__((ext_vector_type(8)))  float    v8f;

#define B_     2
#define N_     256
#define D_EQUI 64
#define D_INV  256
#define D_MSG  128
#define H_     8
#define D_FF   512
#define IN_F   448
#define D_OUTC 136
#define KBLK   64
#define NEG_INF_F (-1000000000.0f)

// workspace byte offsets (all 16B aligned)
#define OFF_QM    0u
#define OFF_KM    (OFF_QM    + 131072u)   // B*N*128 f16
#define OFF_HEADS (OFF_KM    + 131072u)   // B*N*256 f32
#define OFF_SN    (OFF_HEADS + 524288u)   // B*N*64  f32
#define OFF_PROJ  (OFF_SN    + 131072u)   // B*N*192 f32
#define OFF_W1T   (OFF_PROJ  + 393216u)   // 512*448 f16
#define OFF_W2T   (OFF_W1T   + 458752u)   // 144*512 f16
#define OFF_MSG   (OFF_W2T   + 147456u)   // B*N*N*72 f32

// output element offsets
#define EQ_OFF   0
#define INV_OFF  98304
#define EDGE_OFF 229376

static __device__ __forceinline__ v16h comb16(v8h lo, v8h hi) {
    union { v16h v; v8h h[2]; } u;
    u.h[0] = lo; u.h[1] = hi;
    return u.v;
}

// ---------------------------------------------------------------- kernel 1 --
__global__ __launch_bounds__(256) void prep_weights(
    const float* __restrict__ W1, const float* __restrict__ W2,
    _Float16* __restrict__ w1t, _Float16* __restrict__ w2t) {
    int tid = blockIdx.x * blockDim.x + threadIdx.x;
    int stride = gridDim.x * blockDim.x;
    for (int idx = tid; idx < D_FF * IN_F; idx += stride) {
        int n = idx / IN_F, k = idx % IN_F;
        w1t[idx] = (_Float16)W1[k * D_FF + n];          // W1t[n][k] = W1[k][n]
    }
    for (int idx = tid; idx < 144 * D_FF; idx += stride) {
        int n = idx / D_FF, k = idx % D_FF;
        w2t[idx] = (n < D_OUTC) ? (_Float16)W2[k * D_OUTC + n] : (_Float16)0.0f;
    }
}

// ---------------------------------------------------------------- kernel 2 --
__global__ __launch_bounds__(256) void precompute_nodes(
    const float* __restrict__ equis, const float* __restrict__ invs,
    const float* __restrict__ Wq, const float* __restrict__ bq,
    const float* __restrict__ Wk, const float* __restrict__ bk,
    const float* __restrict__ Win, const float* __restrict__ b_in,
    const float* __restrict__ Wcoord,
    _Float16* __restrict__ qm, _Float16* __restrict__ km,
    float* __restrict__ heads, float* __restrict__ sn, float* __restrict__ proj) {
    __shared__ float invrow[D_INV];
    __shared__ float eqrow[3 * D_EQUI];
    int bn = blockIdx.x, t = threadIdx.x;
    invrow[t] = invs[(size_t)bn * D_INV + t];
    if (t < 192) eqrow[t] = equis[(size_t)bn * 192 + t];
    __syncthreads();

    if (t < D_MSG) {
        float aq = bq[t], ak = bk[t];
        for (int i = 0; i < D_INV; i++) {
            float v = invrow[i];
            aq += v * Wq[i * D_MSG + t];
            ak += v * Wk[i * D_MSG + t];
        }
        qm[(size_t)bn * D_MSG + t] = (_Float16)aq;
        km[(size_t)bn * D_MSG + t] = (_Float16)ak;
    }
    {
        float ah = b_in[t];
        for (int i = 0; i < D_INV; i++) ah += invrow[i] * Win[i * D_INV + t];
        heads[(size_t)bn * D_INV + t] = ah;
    }
    if (t < D_EQUI) {
        float x = eqrow[t], y = eqrow[64 + t], z = eqrow[128 + t];
        sn[(size_t)bn * D_EQUI + t] = x * x + y * y + z * z;
    }
    if (t < 192) {
        int c = t >> 6, e = t & 63;
        float p = 0.f;
        for (int d = 0; d < 64; d++) p += eqrow[c * 64 + d] * Wcoord[d * 64 + e];
        proj[(size_t)bn * 192 + t] = p;
    }
}

// ---------------------------------------------------------------- kernel 3 --
// one WG per (b, q, k-block of 64): builds pf tile, WMMA MLP, routes outputs
__global__ __launch_bounds__(256) void edge_mlp(
    const float* __restrict__ equis, const float* __restrict__ edges,
    const float* __restrict__ b1, const float* __restrict__ b2,
    const _Float16* __restrict__ qm, const _Float16* __restrict__ km,
    const float* __restrict__ sn,
    const _Float16* __restrict__ w1t, const _Float16* __restrict__ w2t,
    float* __restrict__ msgs72, float* __restrict__ edge_out) {
    __shared__ __align__(16) _Float16 pf[KBLK][456];   // 448 + pad
    __shared__ __align__(16) _Float16 hid[KBLK][520];  // 512 + pad
    __shared__ float b1s[D_FF];
    __shared__ float b2s[144];
    __shared__ float eqq[192];
    __shared__ float snq[64];
    __shared__ _Float16 qms[128];

    int t  = threadIdx.x;
    int kb = blockIdx.x & 3;
    int q  = (blockIdx.x >> 2) & 255;
    int b  = blockIdx.x >> 10;
    int bq = b * N_ + q;
    int k0 = kb * KBLK;

    if (t < 128) qms[t] = qm[(size_t)bq * 128 + t];
    for (int i = t; i < D_FF; i += 256) b1s[i] = b1[i];
    if (t < 144) b2s[t] = (t < D_OUTC) ? b2[t] : 0.f;
    if (t < 192) eqq[t] = equis[(size_t)bq * 192 + t];
    if (t < 64)  snq[t] = sn[(size_t)bq * 64 + t];
    __syncthreads();

    // ---- assemble pf = [q_msg | k_msg | dot | dist | edges] in f16 --------
    for (int idx = t; idx < KBLK * 128; idx += 256) {
        int r = idx >> 7, c = idx & 127;
        pf[r][c] = qms[c];
    }
    for (int idx = t; idx < KBLK * 128; idx += 256) {
        int r = idx >> 7, c = idx & 127;
        pf[r][128 + c] = km[((size_t)(b * N_ + k0 + r)) * 128 + c];
    }
    for (int idx = t; idx < KBLK * 64; idx += 256) {
        int r = idx >> 6, d = idx & 63;
        int kn = b * N_ + k0 + r;
        const float* ek = equis + (size_t)kn * 192;
        float dotv = eqq[d] * ek[d] + eqq[64 + d] * ek[64 + d] + eqq[128 + d] * ek[128 + d];
        float sq = snq[d] + sn[(size_t)kn * 64 + d] - 2.f * dotv;
        sq = fmaxf(sq, 0.f);
        float dist = (sq > 0.f) ? sqrtf(sq) : 0.f;
        pf[r][256 + d] = (_Float16)dotv;
        pf[r][320 + d] = (_Float16)dist;
    }
    for (int idx = t; idx < KBLK * 64; idx += 256) {
        int r = idx >> 6, d = idx & 63;
        pf[r][384 + d] = (_Float16)edges[((size_t)bq * N_ + k0 + r) * 64 + d];
    }
    __syncthreads();

    int wave  = t >> 5, lane = t & 31;
    int mrow  = lane & 15;            // row (A) / col (B) within tile
    int khalf = (lane >> 4) * 8;      // K-chunk select per ISA 16-bit layout

    // ---- GEMM1: (64 x 448) @ (448 x 512), f32 accum, 16-way ILP ------------
    v8f acc[4][4] = {};
    for (int kk = 0; kk < 14; kk++) {
        int kc = kk * 32;
        v16h afr[4];
        for (int mt = 0; mt < 4; mt++) {
            int m = mt * 16 + mrow;
            afr[mt] = comb16(*(const v8h*)&pf[m][kc + khalf],
                             *(const v8h*)&pf[m][kc + khalf + 16]);
        }
        v16h bfr[4];
        for (int nt = 0; nt < 4; nt++) {
            const _Float16* wr = w1t + (size_t)((wave * 4 + nt) * 16 + mrow) * IN_F + kc + khalf;
            bfr[nt] = comb16(*(const v8h*)wr, *(const v8h*)(wr + 16));
        }
        for (int mt = 0; mt < 4; mt++)
            for (int nt = 0; nt < 4; nt++)
                acc[mt][nt] = __builtin_amdgcn_wmma_f32_16x16x32_f16(
                    false, afr[mt], false, bfr[nt], (short)0, acc[mt][nt], false, false);
    }
    // SiLU epilogue -> hid (f16) ; D layout: m = r + 8*(lane>>4), n = lane&15
    for (int mt = 0; mt < 4; mt++)
        for (int nt = 0; nt < 4; nt++) {
            int ncol  = (wave * 4 + nt) * 16 + mrow;
            int mbase = mt * 16 + (lane >> 4) * 8;
            float bb = b1s[ncol];
            for (int r = 0; r < 8; r++) {
                float x = acc[mt][nt][r] + bb;
                float s = x / (1.f + __expf(-x));
                hid[mbase + r][ncol] = (_Float16)s;
            }
        }
    __syncthreads();

    // ---- GEMM2: (64 x 512) @ (512 x 144pad), two jobs per wave for ILP ----
    for (int jb = wave; jb < 36; jb += 16) {
        int j0 = jb, j1 = jb + 8;
        bool has1 = (j1 < 36);                 // wave-uniform
        int mt0 = j0 / 9, nt0 = j0 % 9;
        int mt1 = has1 ? (j1 / 9) : 0;
        int nt1 = has1 ? (j1 % 9) : 0;
        v8f a0 = {}, a1 = {};
        for (int kk = 0; kk < 16; kk++) {
            int kc = kk * 32;
            int m0 = mt0 * 16 + mrow;
            v16h afr0 = comb16(*(const v8h*)&hid[m0][kc + khalf],
                               *(const v8h*)&hid[m0][kc + khalf + 16]);
            const _Float16* wr0 = w2t + (size_t)(nt0 * 16 + mrow) * D_FF + kc + khalf;
            v16h bfr0 = comb16(*(const v8h*)wr0, *(const v8h*)(wr0 + 16));
            if (has1) {
                int m1 = mt1 * 16 + mrow;
                v16h afr1 = comb16(*(const v8h*)&hid[m1][kc + khalf],
                                   *(const v8h*)&hid[m1][kc + khalf + 16]);
                const _Float16* wr1 = w2t + (size_t)(nt1 * 16 + mrow) * D_FF + kc + khalf;
                v16h bfr1 = comb16(*(const v8h*)wr1, *(const v8h*)(wr1 + 16));
                a0 = __builtin_amdgcn_wmma_f32_16x16x32_f16(
                    false, afr0, false, bfr0, (short)0, a0, false, false);
                a1 = __builtin_amdgcn_wmma_f32_16x16x32_f16(
                    false, afr1, false, bfr1, (short)0, a1, false, false);
            } else {
                a0 = __builtin_amdgcn_wmma_f32_16x16x32_f16(
                    false, afr0, false, bfr0, (short)0, a0, false, false);
            }
        }
        // write both jobs
        for (int s = 0; s < 2; s++) {
            if (s == 1 && !has1) break;
            int mt = s ? mt1 : mt0, nt2 = s ? nt1 : nt0;
            v8f av = s ? a1 : a0;
            int ncol  = nt2 * 16 + mrow;
            int mbase = mt * 16 + (lane >> 4) * 8;
            float bb = b2s[ncol];
            for (int r = 0; r < 8; r++) {
                int kidx = k0 + mbase + r;
                float v = av[r] + bb;
                if (ncol < 72)
                    msgs72[((size_t)bq * N_ + kidx) * 72 + ncol] = v;          // attn logits
                else if (ncol < D_OUTC)
                    edge_out[((size_t)bq * N_ + kidx) * 64 + (ncol - 72)] = v; // edge_feats
            }
        }
    }
}

// ---------------------------------------------------------------- kernel 4 --
// one WG per (b,q): masked softmax over k (72 cols) + aggregations
__global__ __launch_bounds__(256) void aggregate(
    const float* __restrict__ msgs72, const int* __restrict__ adj,
    const float* __restrict__ heads, const float* __restrict__ proj,
    const float* __restrict__ Wout, const float* __restrict__ b_out,
    const float* __restrict__ Wattn, float* __restrict__ out) {
    __shared__ float mg[N_][73];
    __shared__ float maskadd[N_];
    __shared__ float wv[72];
    __shared__ float sinv[256];
    __shared__ float eqo[192];
    int t = threadIdx.x;
    int bq = blockIdx.x;
    int b = bq >> 8;

    for (int idx = t; idx < N_ * 72; idx += 256) {
        int k = idx / 72, j = idx % 72;
        mg[k][j] = msgs72[((size_t)bq * N_ + k) * 72 + j];
    }
    maskadd[t] = (adj[(size_t)bq * N_ + t] > 0) ? 0.f : NEG_INF_F;
    __syncthreads();

    if (t < 72) {
        float mx = -3.4e38f;
        for (int k = 0; k < N_; k++) mx = fmaxf(mx, mg[k][t] + maskadd[k]);
        float s = 0.f;
        for (int k = 0; k < N_; k++) { float e = __expf(mg[k][t] + maskadd[k] - mx); mg[k][t] = e; s += e; }
        float inv = 1.f / s;
        float ss = 0.f;
        for (int k = 0; k < N_; k++) { float a = mg[k][t] * inv; mg[k][t] = a; ss += a * a; }
        wv[t] = sqrtf(ss);
    }
    __syncthreads();

    {   // inv_out[h,d] = sum_k a_inv[k,h] * heads[b,k,h*32+d], scaled by w_inv[h]
        int h = t >> 5;
        const float* hp = heads + (size_t)b * N_ * D_INV + t;
        float o = 0.f;
        for (int k = 0; k < N_; k++) o += mg[k][h] * hp[(size_t)k * D_INV];
        sinv[t] = o * wv[h];
    }
    if (t < 192) {  // eq_out[c,d] = sum_k a_eq[k,d] * proj[b,k,c,d], scaled by w_eq[d]
        int d = t & 63;
        const float* pp = proj + (size_t)b * N_ * 192 + t;
        float o = 0.f;
        for (int k = 0; k < N_; k++) o += mg[k][8 + d] * pp[(size_t)k * 192];
        eqo[t] = o * wv[8 + d];
    }
    __syncthreads();

    {   // inv_updates = sinv @ Wout + b_out
        float r = b_out[t];
        for (int i = 0; i < D_INV; i++) r += sinv[i] * Wout[i * D_INV + t];
        out[INV_OFF + (size_t)bq * D_INV + t] = r;
    }
    if (t < 192) {  // equi_updates = eq_out @ Wattn
        int c = t >> 6, e = t & 63;
        float r = 0.f;
        for (int d = 0; d < 64; d++) r += eqo[c * 64 + d] * Wattn[d * 64 + e];
        out[EQ_OFF + (size_t)bq * 192 + t] = r;
    }
}

// ------------------------------------------------------------------ launch --
extern "C" void kernel_launch(void* const* d_in, const int* in_sizes, int n_in,
                              void* d_out, int out_size, void* d_ws, size_t ws_size,
                              hipStream_t stream) {
    const float* equis  = (const float*)d_in[0];
    const float* invs   = (const float*)d_in[1];
    const float* edges  = (const float*)d_in[2];
    const int*   adj    = (const int*)d_in[3];
    const float* Wq     = (const float*)d_in[4];
    const float* bq     = (const float*)d_in[5];
    const float* Wk     = (const float*)d_in[6];
    const float* bk     = (const float*)d_in[7];
    const float* W1     = (const float*)d_in[8];
    const float* b1     = (const float*)d_in[9];
    const float* W2     = (const float*)d_in[10];
    const float* b2     = (const float*)d_in[11];
    const float* Win    = (const float*)d_in[12];
    const float* b_in   = (const float*)d_in[13];
    const float* Wout   = (const float*)d_in[14];
    const float* b_out  = (const float*)d_in[15];
    const float* Wcoord = (const float*)d_in[16];
    const float* Wattn  = (const float*)d_in[17];
    float* out = (float*)d_out;

    char* ws = (char*)d_ws;
    _Float16* qm    = (_Float16*)(ws + OFF_QM);
    _Float16* km    = (_Float16*)(ws + OFF_KM);
    float*    heads = (float*)(ws + OFF_HEADS);
    float*    sn    = (float*)(ws + OFF_SN);
    float*    proj  = (float*)(ws + OFF_PROJ);
    _Float16* w1t   = (_Float16*)(ws + OFF_W1T);
    _Float16* w2t   = (_Float16*)(ws + OFF_W2T);
    float*    msgs  = (float*)(ws + OFF_MSG);

    prep_weights<<<256, 256, 0, stream>>>(W1, W2, w1t, w2t);
    precompute_nodes<<<B_ * N_, 256, 0, stream>>>(equis, invs, Wq, bq, Wk, bk,
                                                  Win, b_in, Wcoord,
                                                  qm, km, heads, sn, proj);
    edge_mlp<<<B_ * N_ * (N_ / KBLK), 256, 0, stream>>>(equis, edges, b1, b2,
                                                        qm, km, sn, w1t, w2t,
                                                        msgs, out + EDGE_OFF);
    aggregate<<<B_ * N_, 256, 0, stream>>>(msgs, adj, heads, proj,
                                           Wout, b_out, Wattn, out);
}
